// PointNetSetAbstraction_82755429859696
// MI455X (gfx1250) — matrix operand order repaired and checked by reading, not compiled, not run on hardware
//
#include <hip/hip_runtime.h>
#include <hip/hip_bf16.h>

#define NPTS    4096
#define NBATCH  16
#define NPOINT  512
#define NSAMPLE 32
#define CIN     64
#define RADIUS2 0.04f
#define BN_EPS  1e-5f

typedef __attribute__((ext_vector_type(16))) _Float16 v16h;
typedef __attribute__((ext_vector_type(8)))  _Float16 v8h;
typedef __attribute__((ext_vector_type(8)))  float    v8f;

// ---------------------------------------------------------------------------
// Prep: fold BN (inference) into conv weights -> f16 W', f32 b' in workspace
// ---------------------------------------------------------------------------
__global__ __launch_bounds__(256)
void prep_kernel(const float* __restrict__ w0, const float* __restrict__ b0,
                 const float* __restrict__ g0, const float* __restrict__ be0,
                 const float* __restrict__ rm0, const float* __restrict__ rv0,
                 const float* __restrict__ w1, const float* __restrict__ b1,
                 const float* __restrict__ g1, const float* __restrict__ be1,
                 const float* __restrict__ rm1, const float* __restrict__ rv1,
                 const float* __restrict__ w2, const float* __restrict__ b2,
                 const float* __restrict__ g2, const float* __restrict__ be2,
                 const float* __restrict__ rm2, const float* __restrict__ rv2,
                 _Float16* __restrict__ W0, _Float16* __restrict__ W1,
                 _Float16* __restrict__ W2,
                 float* __restrict__ B0, float* __restrict__ B1,
                 float* __restrict__ B2) {
  const int tid = blockIdx.x * blockDim.x + threadIdx.x;
  const int nth = gridDim.x * blockDim.x;
  // W0: 64 x 67, pad cin to 96
  for (int i = tid; i < 64 * 96; i += nth) {
    const int o = i / 96, c = i % 96;
    const float sc = g0[o] * rsqrtf(rv0[o] + BN_EPS);
    W0[i] = (c < 67) ? (_Float16)(w0[o * 67 + c] * sc) : (_Float16)0.0f;
  }
  for (int i = tid; i < 64 * 64; i += nth) {
    const int o = i >> 6, c = i & 63;
    const float sc = g1[o] * rsqrtf(rv1[o] + BN_EPS);
    W1[i] = (_Float16)(w1[o * 64 + c] * sc);
  }
  for (int i = tid; i < 128 * 64; i += nth) {
    const int o = i >> 6, c = i & 63;
    const float sc = g2[o] * rsqrtf(rv2[o] + BN_EPS);
    W2[i] = (_Float16)(w2[o * 64 + c] * sc);
  }
  if (tid < 64) {
    const float sc = g0[tid] * rsqrtf(rv0[tid] + BN_EPS);
    B0[tid] = (b0[tid] - rm0[tid]) * sc + be0[tid];
  }
  if (tid < 64) {
    const float sc = g1[tid] * rsqrtf(rv1[tid] + BN_EPS);
    B1[tid] = (b1[tid] - rm1[tid]) * sc + be1[tid];
  }
  if (tid < 128) {
    const float sc = g2[tid] * rsqrtf(rv2[tid] + BN_EPS);
    B2[tid] = (b2[tid] - rm2[tid]) * sc + be2[tid];
  }
}

// ---------------------------------------------------------------------------
// Farthest point sampling: one workgroup per batch; xyz + distances in LDS
// ---------------------------------------------------------------------------
__global__ __launch_bounds__(256)
void fps_kernel(const float* __restrict__ xyz, float* __restrict__ new_xyz) {
  const int b = blockIdx.x;
  const int tid = threadIdx.x;
  __shared__ float lx[NPTS], ly[NPTS], lz[NPTS];
  __shared__ float dist[NPTS];
  __shared__ float rv[256];
  __shared__ int   ri[256];
  __shared__ int   cur;
  const float* X = xyz + (size_t)b * NPTS * 3;
  for (int i = tid; i < NPTS; i += 256) {
    lx[i] = X[i * 3 + 0];
    ly[i] = X[i * 3 + 1];
    lz[i] = X[i * 3 + 2];
    dist[i] = 1e10f;
  }
  if (tid == 0) cur = 0;
  __syncthreads();
  for (int it = 0; it < NPOINT; ++it) {
    const int far = cur;
    const float cx = lx[far], cy = ly[far], cz = lz[far];
    if (tid == 0) {
      float* o = new_xyz + ((size_t)b * NPOINT + it) * 3;
      o[0] = cx; o[1] = cy; o[2] = cz;
    }
    float best = -1.0f; int bi = 0;
    for (int i = tid; i < NPTS; i += 256) {
      const float dx = lx[i] - cx, dy = ly[i] - cy, dz = lz[i] - cz;
      const float d = dx * dx + dy * dy + dz * dz;
      const float nd = fminf(dist[i], d);
      dist[i] = nd;
      if (nd > best) { best = nd; bi = i; }
    }
    rv[tid] = best; ri[tid] = bi;
    __syncthreads();
    for (int s = 128; s > 0; s >>= 1) {
      if (tid < s && rv[tid + s] > rv[tid]) { rv[tid] = rv[tid + s]; ri[tid] = ri[tid + s]; }
      __syncthreads();
    }
    if (tid == 0) cur = ri[0];
    __syncthreads();
  }
}

// ---------------------------------------------------------------------------
// WMMA fragment loaders (16-bit A 16x32 / B 32x16, both in contiguous k = 16*half + 0..15 order)
// ---------------------------------------------------------------------------
__device__ __forceinline__ v16h load_a_frag(const _Float16* __restrict__ W,
                                            int ldw, int ot, int kc, int lane) {
  const int m = lane & 15, hi = lane >> 4;
  const _Float16* p = W + (size_t)(ot * 16 + m) * ldw + kc * 32 + hi * 16;
  return *(const v16h*)p;           // K = hi*16 + 0..15
}

__device__ __forceinline__ v16h load_b_frag(const _Float16* Xin, int ldx,
                                            int kc, int nt, int lane) {
  const int n = lane & 15, hiK = lane >> 4;
  const _Float16* p = Xin + (nt * 16 + n) * ldx + kc * 32 + hiK * 16;
  return *(const v16h*)p;           // K = hiK*16 + 0..15, two ds_load_b128
}

template <int OT, int KC>
__device__ __forceinline__ void mlp_layer(const _Float16* __restrict__ W, int ldw,
                                          const float* __restrict__ Bb,
                                          const _Float16* Xin, int ldin,
                                          _Float16* Yout, int ldout, int lane) {
  const int n = lane & 15;
  const int off = (lane >> 4) * 8;
  for (int ot = 0; ot < OT; ++ot) {
    for (int nt = 0; nt < 2; ++nt) {
      v8f acc = {};
#pragma unroll
      for (int kc = 0; kc < KC; ++kc) {
        const v16h a = load_a_frag(W, ldw, ot, kc, lane);
        const v16h bf = load_b_frag(Xin, ldin, kc, nt, lane);
        acc = __builtin_amdgcn_wmma_f32_16x16x32_f16(false, a, false, bf,
                                                     (short)0, acc, false, false);
      }
      v8h y;
#pragma unroll
      for (int r = 0; r < 8; ++r) {
        float v = acc[r] + Bb[ot * 16 + off + r];
        v = v > 0.0f ? v : 0.0f;
        y[r] = (_Float16)v;
      }
      *(v8h*)(Yout + (nt * 16 + n) * ldout + ot * 16 + off) = y;
    }
  }
}

// ---------------------------------------------------------------------------
// Per-(b,s): ball query + gather + 3-layer WMMA MLP + max over K
// ---------------------------------------------------------------------------
__global__ __launch_bounds__(32)
void group_mlp_kernel(const float* __restrict__ xyz,
                      const float* __restrict__ points,
                      const float* __restrict__ new_xyz,
                      const _Float16* __restrict__ W0, const float* __restrict__ B0,
                      const _Float16* __restrict__ W1, const float* __restrict__ B1,
                      const _Float16* __restrict__ W2, const float* __restrict__ B2,
                      float* __restrict__ out_pts) {
  const int s = blockIdx.x;
  const int b = blockIdx.y;
  const int lane = threadIdx.x;

  __shared__ _Float16 x0[NSAMPLE * 96];  // layer-0 input, cin padded 67->96
  __shared__ _Float16 y0[NSAMPLE * 64];
  __shared__ _Float16 y1[NSAMPLE * 64];
  __shared__ int gidx[NSAMPLE];

  __builtin_prefetch(W0, 0, 1);
  __builtin_prefetch(W1, 0, 1);
  __builtin_prefetch(W2, 0, 1);

  const float* X = xyz + (size_t)b * NPTS * 3;
  const float* c = new_xyz + ((size_t)b * NPOINT + s) * 3;
  const float cx = c[0], cy = c[1], cz = c[2];

  // --- ball query: wave32 ballot compaction, ordered by index ---
  int cnt = 0;
  for (int base = 0; base < NPTS && cnt < NSAMPLE; base += 32) {
    const int i = base + lane;
    const float dx = X[i * 3 + 0] - cx;
    const float dy = X[i * 3 + 1] - cy;
    const float dz = X[i * 3 + 2] - cz;
    const float d2 = dx * dx + dy * dy + dz * dz;
    const bool in = d2 <= RADIUS2;
    const unsigned m = (unsigned)__ballot(in);
    const int pos = cnt + __popc(m & ((1u << lane) - 1u));
    if (in && pos < NSAMPLE) gidx[pos] = i;
    cnt += __popc(m);
  }
  __syncthreads();
  if (cnt > NSAMPLE) cnt = NSAMPLE;
  const int fb = (cnt > 0) ? gidx[0] : 0;
  if (lane >= cnt) gidx[lane] = fb;   // group_first fallback
  __syncthreads();

  // --- gather: [xyz - centroid (3) || points (64)] -> f16 LDS, pad to 96 ---
  {
    const int j = gidx[lane];
    _Float16* row = x0 + lane * 96;
    row[0] = (_Float16)(X[j * 3 + 0] - cx);
    row[1] = (_Float16)(X[j * 3 + 1] - cy);
    row[2] = (_Float16)(X[j * 3 + 2] - cz);
    const float* pt = points + ((size_t)b * NPTS + j) * CIN;
#pragma unroll
    for (int cc = 0; cc < CIN; cc += 4) {
      const float4 v = *(const float4*)(pt + cc);
      row[3 + cc + 0] = (_Float16)v.x;
      row[3 + cc + 1] = (_Float16)v.y;
      row[3 + cc + 2] = (_Float16)v.z;
      row[3 + cc + 3] = (_Float16)v.w;
    }
#pragma unroll
    for (int cc = 67; cc < 96; ++cc) row[cc] = (_Float16)0.0f;
  }
  __syncthreads();

  mlp_layer<4, 3>(W0, 96, B0, x0, 96, y0, 64, lane);  // 67->64
  __syncthreads();
  mlp_layer<4, 2>(W1, 64, B1, y0, 64, y1, 64, lane);  // 64->64
  __syncthreads();

  // --- layer 2 (64->128) fused with max over K=32 ---
  const int off = (lane >> 4) * 8;
  float* obase = out_pts + ((size_t)b * NPOINT + s) * 128;
  for (int ot = 0; ot < 8; ++ot) {
    v8f a0 = {}, a1 = {};
#pragma unroll
    for (int kc = 0; kc < 2; ++kc) {
      const v16h a = load_a_frag(W2, 64, ot, kc, lane);
      const v16h bf0 = load_b_frag(y1, 64, kc, 0, lane);
      const v16h bf1 = load_b_frag(y1, 64, kc, 1, lane);
      a0 = __builtin_amdgcn_wmma_f32_16x16x32_f16(false, a, false, bf0,
                                                  (short)0, a0, false, false);
      a1 = __builtin_amdgcn_wmma_f32_16x16x32_f16(false, a, false, bf1,
                                                  (short)0, a1, false, false);
    }
    float m[8];
#pragma unroll
    for (int r = 0; r < 8; ++r) {
      const float v = fmaxf(a0[r], a1[r]) + B2[ot * 16 + off + r];
      m[r] = fmaxf(v, 0.0f);
    }
    // max over the 16 positions held across each 16-lane half
#pragma unroll
    for (int d = 1; d < 16; d <<= 1) {
#pragma unroll
      for (int r = 0; r < 8; ++r) m[r] = fmaxf(m[r], __shfl_xor(m[r], d, 32));
    }
    if ((lane & 15) == 0) {
#pragma unroll
      for (int r = 0; r < 8; ++r) obase[ot * 16 + off + r] = m[r];
    }
  }
}

// ---------------------------------------------------------------------------
extern "C" void kernel_launch(void* const* d_in, const int* in_sizes, int n_in,
                              void* d_out, int out_size, void* d_ws, size_t ws_size,
                              hipStream_t stream) {
  (void)in_sizes; (void)n_in; (void)out_size; (void)ws_size;
  const float* xyz    = (const float*)d_in[0];
  const float* points = (const float*)d_in[1];
  const float* w0  = (const float*)d_in[2];
  const float* b0  = (const float*)d_in[3];
  const float* g0  = (const float*)d_in[4];
  const float* be0 = (const float*)d_in[5];
  const float* rm0 = (const float*)d_in[6];
  const float* rv0 = (const float*)d_in[7];
  const float* w1  = (const float*)d_in[8];
  const float* b1  = (const float*)d_in[9];
  const float* g1  = (const float*)d_in[10];
  const float* be1 = (const float*)d_in[11];
  const float* rm1 = (const float*)d_in[12];
  const float* rv1 = (const float*)d_in[13];
  const float* w2  = (const float*)d_in[14];
  const float* b2  = (const float*)d_in[15];
  const float* g2  = (const float*)d_in[16];
  const float* be2 = (const float*)d_in[17];
  const float* rm2 = (const float*)d_in[18];
  const float* rv2 = (const float*)d_in[19];

  char* ws = (char*)d_ws;
  _Float16* W0 = (_Float16*)(ws + 0);        // 64*96*2  = 12288
  _Float16* W1 = (_Float16*)(ws + 12288);    // 64*64*2  =  8192
  _Float16* W2 = (_Float16*)(ws + 20480);    // 128*64*2 = 16384
  float*    B0 = (float*)(ws + 36864);       // 256
  float*    B1 = (float*)(ws + 37120);       // 256
  float*    B2 = (float*)(ws + 37376);       // 512

  float* out     = (float*)d_out;
  float* out_xyz = out;                          // (B, 512, 3)
  float* out_pts = out + NBATCH * NPOINT * 3;    // (B, 512, 128)

  prep_kernel<<<32, 256, 0, stream>>>(w0, b0, g0, be0, rm0, rv0,
                                      w1, b1, g1, be1, rm1, rv1,
                                      w2, b2, g2, be2, rm2, rv2,
                                      W0, W1, W2, B0, B1, B2);
  fps_kernel<<<NBATCH, 256, 0, stream>>>(xyz, out_xyz);
  dim3 grid(NPOINT, NBATCH);
  group_mlp_kernel<<<grid, 32, 0, stream>>>(xyz, points, out_xyz,
                                            W0, B0, W1, B1, W2, B2, out_pts);
}
